// EquivariantOutputHead_9457517986018
// MI455X (gfx1250) — matrix operand order repaired and checked by reading, not compiled, not run on hardware
//
#include <hip/hip_runtime.h>

typedef __attribute__((ext_vector_type(16))) _Float16 v16h;
typedef __attribute__((ext_vector_type(8)))  float    v8f;

#define BB 8
#define TT 32
#define RR 512
#define DD 256
#define BT (BB*TT)        // 256 (b,t) pairs
#define D2 (DD/2)         // 128
#define W2T_STRIDE (DD+8) // padded K stride for W2^T staging (f16)
#define ROWS_PB 64        // rows per workgroup in main kernel
#define MAIN_THREADS 128  // 4 waves

__device__ __forceinline__ float gelu_tanh(float x) {
    const float c0 = 0.7978845608028654f; // sqrt(2/pi)
    float x3 = x * x * x;
    return 0.5f * x * (1.0f + tanhf(c0 * (x + 0.044715f * x3)));
}

// rotate v by quaternion (w,x,y,z), normalizing like the reference
__device__ __forceinline__ void quat_rotate(float qw, float qx, float qy, float qz,
                                            float vx, float vy, float vz,
                                            float& ox, float& oy, float& oz) {
    float nrm = sqrtf(qw*qw + qx*qx + qy*qy + qz*qz);
    float inv = 1.0f / (nrm + 1e-8f);
    float w = qw*inv, x = qx*inv, y = qy*inv, z = qz*inv;
    ox = (1.f - 2.f*(y*y + z*z))*vx + 2.f*(x*y - w*z)*vy + 2.f*(x*z + w*y)*vz;
    oy = 2.f*(x*y + w*z)*vx + (1.f - 2.f*(x*x + z*z))*vy + 2.f*(y*z - w*x)*vz;
    oz = 2.f*(x*z - w*y)*vx + 2.f*(y*z + w*x)*vy + (1.f - 2.f*(x*x + y*y))*vz;
}

__device__ __forceinline__ void wait_asynccnt0() {
#if __has_builtin(__builtin_amdgcn_s_wait_asynccnt)
    __builtin_amdgcn_s_wait_asynccnt(0);
#else
    asm volatile("s_wait_asynccnt 0x0" ::: "memory");
#endif
}

// ---------------- Kernel 0: pre-convert W2 -> f16, transposed + padded ----------------
__global__ __launch_bounds__(256) void w2cvt_kernel(const float* __restrict__ W2,
                                                    _Float16* __restrict__ w2t_ws) {
    int e = blockIdx.x * 256 + threadIdx.x;      // DD*D2 = 32768 elements
    if (e < DD * D2) {
        int k = e >> 7, n = e & 127;             // W2 is [k][n] row-major
        w2t_ws[n * W2T_STRIDE + k] = (_Float16)W2[e];
    }
}

// ---------------- Kernel 1: per-(b,t) centroid of trans ----------------
__global__ __launch_bounds__(256) void centroid_kernel(const float* __restrict__ trans,
                                                       float* __restrict__ centroid) {
    __shared__ float red[256][3];
    int bt = blockIdx.x;
    int tid = threadIdx.x;
    const float* tb = trans + (size_t)bt * RR * 3;
    float s0 = 0.f, s1 = 0.f, s2 = 0.f;
    for (int r = tid; r < RR; r += 256) {
        s0 += tb[r*3+0]; s1 += tb[r*3+1]; s2 += tb[r*3+2];
    }
    red[tid][0] = s0; red[tid][1] = s1; red[tid][2] = s2;
    __syncthreads();
    for (int st = 128; st > 0; st >>= 1) {
        if (tid < st) {
            red[tid][0] += red[tid+st][0];
            red[tid][1] += red[tid+st][1];
            red[tid][2] += red[tid+st][2];
        }
        __syncthreads();
    }
    if (tid == 0) {
        centroid[bt*3+0] = red[0][0] * (1.0f/RR);
        centroid[bt*3+1] = red[0][1] * (1.0f/RR);
        centroid[bt*3+2] = red[0][2] * (1.0f/RR);
    }
}

// -------- Kernel 2: base1[bt,:] = sf[bt,:] @ W1[0:256,:] + b1 (WMMA) --------
__global__ __launch_bounds__(256) void base1_kernel(const float* __restrict__ sf,
                                                    const float* __restrict__ W1,
                                                    const float* __restrict__ b1,
                                                    float* __restrict__ base1) {
    int wave = threadIdx.x >> 5;
    int lane = threadIdx.x & 31;
    int tile = blockIdx.x * 8 + wave;          // 256 tiles total
    int mt = tile >> 4, nt = tile & 15;
    int l16   = lane & 15;
    int halfg = lane >> 4;
    int mrow = mt * 16 + l16;
    int ncol = nt * 16 + l16;

    v8f acc = {};
    #pragma unroll
    for (int ks = 0; ks < 8; ++ks) {
        int k0 = ks * 32;
        union { v16h v; _Float16 h[16]; } a, b;
        #pragma unroll
        for (int v = 0; v < 8; ++v) {
            int ka = k0 + ((v & 4) ? 16 : 0) + halfg * 8 + (v & 3) * 2;
            a.h[2*v]   = (_Float16)sf[mrow * DD + ka];
            a.h[2*v+1] = (_Float16)sf[mrow * DD + ka + 1];
            int kb = k0 + ((v & 4) ? 16 : 0) + halfg * 8 + (v & 3) * 2;
            b.h[2*v]   = (_Float16)W1[(kb    ) * DD + ncol];
            b.h[2*v+1] = (_Float16)W1[(kb + 1) * DD + ncol];
        }
        acc = __builtin_amdgcn_wmma_f32_16x16x32_f16(false, a.v, false, b.v,
                                                     (short)0, acc, false, false);
    }
    #pragma unroll
    for (int i = 0; i < 8; ++i) {
        int m = mt * 16 + i + 8 * halfg;       // D-matrix row layout
        base1[m * DD + ncol] = acc[i] + b1[ncol];
    }
}

// ---------------- Kernel 3: fused main kernel ----------------
__global__ __launch_bounds__(MAIN_THREADS) void main_kernel(
    const float* __restrict__ quat, const float* __restrict__ trans,
    const float* __restrict__ W1,
    const _Float16* __restrict__ w2t_ws,        // pre-converted f16 W2^T (padded)
    const float* __restrict__ b2, const float* __restrict__ Wt,
    const float* __restrict__ btv, const float* __restrict__ Wr,
    const float* __restrict__ brv, const float* __restrict__ base1,
    const float* __restrict__ centroid, float* __restrict__ out) {

    __shared__ _Float16 h1s[ROWS_PB][DD + 8];   // activations layer1 (f16)
    __shared__ _Float16 w2t[D2][W2T_STRIDE];    // W2 transposed: [n][k] (f16)
    __shared__ float    h2s[ROWS_PB][D2 + 4];   // activations layer2 (f32)
    __shared__ float    base1s[DD];
    __shared__ float    w1rs[3][DD];
    __shared__ float    relps[ROWS_PB][4];
    __shared__ float    wts[D2 * 3], wrs[D2 * 3], b2s[D2];
    __shared__ float    btb[3], brb[3];

    const int tid = threadIdx.x;
    const int btIdx = blockIdx.x >> 3;
    const int rblk  = blockIdx.x & 7;
    const size_t rowG0 = (size_t)btIdx * RR + rblk * ROWS_PB;

    // ---- async-stage W2^T (f16) straight into LDS: 67584 B = 4224 x b128 ----
    {
        unsigned ldsBase = (unsigned)(uintptr_t)(&w2t[0][0]);
        #pragma unroll
        for (int it = 0; it < (D2 * W2T_STRIDE * 2) / (MAIN_THREADS * 16); ++it) {
            unsigned off = (unsigned)(it * MAIN_THREADS + tid) * 16u;
            unsigned lds = ldsBase + off;
            asm volatile("global_load_async_to_lds_b128 %0, %1, %2"
                         :: "v"(lds), "v"(off), "s"(w2t_ws)
                         : "memory");
        }
    }

    // prefetch the per-block quat rows while async copy runs
    __builtin_prefetch(quat + rowG0 * 4, 0, 3);

    // ---- stage per-(b,t) base1 and the 3 rel-pos rows of W1 ----
    for (int e = tid; e < DD; e += MAIN_THREADS) {
        base1s[e]  = base1[btIdx * DD + e];
        w1rs[0][e] = W1[(DD + 0) * DD + e];
        w1rs[1][e] = W1[(DD + 1) * DD + e];
        w1rs[2][e] = W1[(DD + 2) * DD + e];
    }
    for (int e = tid; e < D2 * 3; e += MAIN_THREADS) { wts[e] = Wt[e]; wrs[e] = Wr[e]; }
    if (tid < D2) b2s[tid] = b2[tid];
    if (tid < 3) { btb[tid] = btv[tid]; brb[tid] = brv[tid]; }

    // ---- local_rel_pos = rotate(conj(quat), trans - centroid) ----
    if (tid < ROWS_PB) {
        size_t rg = rowG0 + tid;
        float cx = centroid[btIdx*3+0], cy = centroid[btIdx*3+1], cz = centroid[btIdx*3+2];
        float vx = trans[rg*3+0] - cx, vy = trans[rg*3+1] - cy, vz = trans[rg*3+2] - cz;
        float4 q = *(const float4*)(quat + rg * 4);
        float rx, ry, rz;
        quat_rotate(q.x, -q.y, -q.z, -q.w, vx, vy, vz, rx, ry, rz);  // conjugate rotation
        relps[tid][0] = rx; relps[tid][1] = ry; relps[tid][2] = rz;
    }

    wait_asynccnt0();     // W2^T resident in LDS
    __syncthreads();

    // ---- h1 = gelu(base1 + relpos @ W1[256:259,:]) -> f16 LDS ----
    for (int e = tid; e < ROWS_PB * DD; e += MAIN_THREADS) {
        int r = e >> 8, n = e & 255;
        float v = base1s[n] + relps[r][0]*w1rs[0][n]
                            + relps[r][1]*w1rs[1][n]
                            + relps[r][2]*w1rs[2][n];
        h1s[r][n] = (_Float16)gelu_tanh(v);
    }
    __syncthreads();

    // ---- GEMM2: each wave computes 16 rows x 128 cols, K=256 ----
    const int wave = tid >> 5, lane = tid & 31;
    const int halfg = lane >> 4, l16 = lane & 15;
    const int mrow = wave * 16 + l16;

    v8f acc[8];
    {
        v8f z = {};
        #pragma unroll
        for (int nt = 0; nt < 8; ++nt) acc[nt] = z;
    }
    #pragma unroll
    for (int ks = 0; ks < 8; ++ks) {
        int k0 = ks * 32;
        union { v16h v; unsigned u[8]; } a;
        #pragma unroll
        for (int v = 0; v < 8; ++v) {
            int ka = k0 + ((v & 4) ? 16 : 0) + halfg * 8 + (v & 3) * 2;
            a.u[v] = *(const unsigned*)&h1s[mrow][ka];   // pair contiguous in K
        }
        #pragma unroll
        for (int nt = 0; nt < 8; ++nt) {
            union { v16h v; unsigned u[8]; } b;
            #pragma unroll
            for (int v = 0; v < 8; ++v) {
                int kb = k0 + ((v & 4) ? 16 : 0) + halfg * 8 + (v & 3) * 2;
                b.u[v] = *(const unsigned*)&w2t[nt*16 + l16][kb]; // W2^T: pair contiguous
            }
            acc[nt] = __builtin_amdgcn_wmma_f32_16x16x32_f16(false, a.v, false, b.v,
                                                             (short)0, acc[nt], false, false);
        }
    }
    // ---- bias + gelu -> h2s ----
    #pragma unroll
    for (int nt = 0; nt < 8; ++nt) {
        int n = nt * 16 + l16;
        #pragma unroll
        for (int i = 0; i < 8; ++i) {
            int m = wave * 16 + i + 8 * halfg;
            h2s[m][n] = gelu_tanh(acc[nt][i] + b2s[n]);
        }
    }
    __syncthreads();

    // ---- heads: waves 0-1 = trans head, waves 2-3 = rot head (wave-uniform) ----
    {
        int hr   = tid & 63;
        int head = tid >> 6;
        size_t rg = rowG0 + hr;
        float4 q = *(const float4*)(quat + rg * 4);
        float qw = q.x, qx = q.y, qy = q.z, qz = q.w;
        const float* Wh = head ? wrs : wts;
        float d0 = 0.f, d1 = 0.f, d2 = 0.f;
        #pragma unroll 4
        for (int k = 0; k < D2; ++k) {
            float h = h2s[hr][k];
            d0 += h * Wh[k*3+0];
            d1 += h * Wh[k*3+1];
            d2 += h * Wh[k*3+2];
        }
        float* o = out + rg * 7;
        if (head == 0) {
            float vx = d0 + btb[0], vy = d1 + btb[1], vz = d2 + btb[2];
            float rx, ry, rz;
            quat_rotate(qw, qx, qy, qz, vx, vy, vz, rx, ry, rz);
            o[4] = rx; o[5] = ry; o[6] = rz;
        } else {
            float ox = (d0 + brb[0]) * 0.1f;
            float oy = (d1 + brb[1]) * 0.1f;
            float oz = (d2 + brb[2]) * 0.1f;
            // 0.5 * quat_multiply(quat, [0, ox, oy, oz])
            o[0] = 0.5f * (-qx*ox - qy*oy - qz*oz);
            o[1] = 0.5f * ( qw*ox + qy*oz - qz*oy);
            o[2] = 0.5f * ( qw*oy - qx*oz + qz*ox);
            o[3] = 0.5f * ( qw*oz + qx*oy - qy*ox);
        }
    }
}

extern "C" void kernel_launch(void* const* d_in, const int* in_sizes, int n_in,
                              void* d_out, int out_size, void* d_ws, size_t ws_size,
                              hipStream_t stream) {
    const float* sf    = (const float*)d_in[0];   // (B,T,D)
    const float* quat  = (const float*)d_in[1];   // (B,T,R,4)
    const float* trans = (const float*)d_in[2];   // (B,T,R,3)
    const float* W1    = (const float*)d_in[3];   // (D+3, D)
    const float* b1    = (const float*)d_in[4];   // (D)
    const float* W2    = (const float*)d_in[5];   // (D, D/2)
    const float* b2    = (const float*)d_in[6];   // (D/2)
    const float* Wt    = (const float*)d_in[7];   // (D/2, 3)
    const float* btv   = (const float*)d_in[8];   // (3)
    const float* Wr    = (const float*)d_in[9];   // (D/2, 3)
    const float* brv   = (const float*)d_in[10];  // (3)
    float* out = (float*)d_out;

    float*     ws_base1    = (float*)d_ws;                 // BT*DD floats
    float*     ws_centroid = ws_base1 + BT * DD;           // BT*3 floats (768, 16B-aligned end)
    _Float16*  ws_w2t      = (_Float16*)(ws_centroid + BT * 3 + 64); // padded f16 W2^T

    w2cvt_kernel<<<(DD * D2 + 255) / 256, 256, 0, stream>>>(W2, ws_w2t);
    centroid_kernel<<<BT, 256, 0, stream>>>(trans, ws_centroid);
    base1_kernel<<<BT / 8, 256, 0, stream>>>(sf, W1, b1, ws_base1);
    main_kernel<<<BT * (RR / ROWS_PB), MAIN_THREADS, 0, stream>>>(
        quat, trans, W1, ws_w2t, b2, Wt, btv, Wr, brv, ws_base1, ws_centroid, out);
}
